// GlobalPointerLayer_12128987644495
// MI455X (gfx1250) — matrix operand order repaired and checked by reading, not compiled, not run on hardware
//
#include <hip/hip_runtime.h>
#include <hip/hip_fp16.h>

#define B_     8
#define S_     1024
#define HID_   768
#define HEADS_ 12
#define HS_    64
#define NQK_   (HEADS_ * HS_ * 2)   // 1536
#define M_     (B_ * S_)            // 8192

typedef _Float16 h8  __attribute__((ext_vector_type(8)));
typedef _Float16 h16 __attribute__((ext_vector_type(16)));
typedef float    f8  __attribute__((ext_vector_type(8)));

// masked logit value: finfo(f32).min, then the final /sqrt(64) scaling
#define NEG_MASKED (-3.4028234663852886e38f * 0.125f)

// ---------------------------------------------------------------------------
// f32 -> f16 copy of the activations
// ---------------------------------------------------------------------------
__global__ void cvt_inputs(const float* __restrict__ x,
                           _Float16* __restrict__ y, int n) {
    int i = blockIdx.x * blockDim.x + threadIdx.x;
    if (i < n) y[i] = (_Float16)x[i];
}

// ---------------------------------------------------------------------------
// W [HID][NQK] f32  ->  Wt [NQK][HID] f16 (transposed so B-fragments are
// two contiguous 16B K chunks per lane per the CDNA5 16-bit B-matrix layout)
// ---------------------------------------------------------------------------
__global__ void cvt_w_transpose(const float* __restrict__ W,
                                _Float16* __restrict__ Wt) {
    int i = blockIdx.x * blockDim.x + threadIdx.x;   // over HID_*NQK_
    if (i >= HID_ * NQK_) return;
    int k = i / NQK_;
    int n = i % NQK_;
    Wt[(size_t)n * HID_ + k] = (_Float16)W[i];
}

// ---------------------------------------------------------------------------
// Projection GEMM: H[M][NQK] = A[M][HID] x Wt^T + bias, f16 in, f32 acc,
// f16 out.  8 waves / block; each wave owns a 16x64 tile (4 WMMA accums),
// K-loop of HID/32 = 24 v_wmma_f32_16x16x32_f16 steps per accumulator.
// ---------------------------------------------------------------------------
__global__ void __launch_bounds__(256)
proj_gemm(const _Float16* __restrict__ A, const _Float16* __restrict__ Wt,
          const float* __restrict__ bias, _Float16* __restrict__ H) {
    const int wave    = threadIdx.x >> 5;
    const int lane    = threadIdx.x & 31;
    const int half_id = lane >> 4;     // which half-wave (K-split of A/B frag)
    const int ln      = lane & 15;
    const int m0      = blockIdx.x * 128 + wave * 16;
    const int n0      = blockIdx.y * 64;

    f8 acc[4] = {};
    const _Float16* arow = A + (size_t)(m0 + ln) * HID_ + half_id * 8;

    for (int k = 0; k < HID_; k += 32) {
        // A fragment, 16-bit 16x32 layout: lane holds two contiguous 8-half
        // chunks (K = base..base+7 and base+16..base+23, base = half_id*8)
        union { h16 v; h8 p[2]; } a;
        a.p[0] = *(const h8*)(arow + k);
        a.p[1] = *(const h8*)(arow + k + 16);
#pragma unroll
        for (int t = 0; t < 4; ++t) {
            // B fragment: lane = column n, two 8-half K chunks (base, base+16)
            const _Float16* brow =
                Wt + (size_t)(n0 + t * 16 + ln) * HID_ + k + half_id * 8;
            union { h16 v; h8 p[2]; } bu;
            bu.p[0] = *(const h8*)brow;
            bu.p[1] = *(const h8*)(brow + 16);
            h16 b = bu.v;
            acc[t] = __builtin_amdgcn_wmma_f32_16x16x32_f16(
                false, a.v, false, b, (short)0, acc[t], false, false);
        }
    }

#pragma unroll
    for (int t = 0; t < 4; ++t) {
        int   col = n0 + t * 16 + ln;
        float bv  = bias[col];
#pragma unroll
        for (int r = 0; r < 8; ++r) {
            int row = m0 + half_id * 8 + r;   // C/D layout: VGPR r -> M=r(+8)
            H[(size_t)row * NQK_ + col] = (_Float16)(acc[t][r] + bv);
        }
    }
}

// ---------------------------------------------------------------------------
// RoPE + Q/K split.  One thread per (m, head, pair): the rotated pair
// (2i, 2i+1) is adjacent in memory so no cross-lane traffic is needed.
// Writes Q/K as [b][head][s][d] row-major for the attention GEMM fragments.
// ---------------------------------------------------------------------------
__global__ void rope_split(const _Float16* __restrict__ H,
                           _Float16* __restrict__ Q,
                           _Float16* __restrict__ Kv) {
    const int PAIRS = HS_ / 2;  // 32
    int idx = blockIdx.x * blockDim.x + threadIdx.x;
    if (idx >= M_ * HEADS_ * PAIRS) return;

    int i    = idx % PAIRS;
    int head = (idx / PAIRS) % HEADS_;
    int m    = idx / (PAIRS * HEADS_);
    int p    = m % S_;                 // sequence position

    float theta = powf(10000.0f, (-2.0f * (float)i) / (float)HS_);
    float ang   = (float)p * theta;
    float sn    = sinf(ang);
    float cs    = cosf(ang);

    const _Float16* hrow = H + (size_t)m * NQK_ + head * (2 * HS_);
    float q0 = (float)hrow[2 * i];
    float q1 = (float)hrow[2 * i + 1];
    float k0 = (float)hrow[HS_ + 2 * i];
    float k1 = (float)hrow[HS_ + 2 * i + 1];

    size_t base =
        (((size_t)(m / S_) * HEADS_ + head) * S_ + p) * HS_ + 2 * i;
    Q[base]      = (_Float16)(q0 * cs - q1 * sn);
    Q[base + 1]  = (_Float16)(q1 * cs + q0 * sn);
    Kv[base]     = (_Float16)(k0 * cs - k1 * sn);
    Kv[base + 1] = (_Float16)(k1 * cs + k0 * sn);
}

// ---------------------------------------------------------------------------
// logits[bh][m][n] = (Q[bh][m] . K[bh][n]) / 8, causal+pad masked.
// Each wave computes a 16x64 tile; tiles fully below the diagonal skip the
// WMMA work (wave-uniform branch, so EXEC stays all-ones around WMMA) and
// just stream out the masked constant.
// ---------------------------------------------------------------------------
__global__ void __launch_bounds__(256)
attn_logits(const _Float16* __restrict__ Q, const _Float16* __restrict__ Kv,
            const unsigned char* __restrict__ mask, float* __restrict__ out) {
    const int wave    = threadIdx.x >> 5;
    const int lane    = threadIdx.x & 31;
    const int half_id = lane >> 4;
    const int ln      = lane & 15;
    const int bh      = blockIdx.x;             // 0..95
    const int b       = bh / HEADS_;
    const int m0      = blockIdx.y * 16;
    const int n0      = blockIdx.z * 512 + wave * 64;

    float* obase = out + ((size_t)bh * S_ + m0) * S_;

    if (n0 + 63 < m0) {  // strictly below diagonal: whole tile masked
#pragma unroll
        for (int t = 0; t < 4; ++t) {
            int col = n0 + t * 16 + ln;
#pragma unroll
            for (int r = 0; r < 8; ++r) {
                int row = r + half_id * 8;
                obase[(size_t)row * S_ + col] = NEG_MASKED;
            }
        }
        return;
    }

    f8 acc[4] = {};
    const _Float16* qrow =
        Q + ((size_t)bh * S_ + (m0 + ln)) * HS_ + half_id * 8;
#pragma unroll
    for (int ks = 0; ks < HS_; ks += 32) {
        union { h16 v; h8 p[2]; } a;
        a.p[0] = *(const h8*)(qrow + ks);
        a.p[1] = *(const h8*)(qrow + ks + 16);
#pragma unroll
        for (int t = 0; t < 4; ++t) {
            const _Float16* krow =
                Kv + ((size_t)bh * S_ + (n0 + t * 16 + ln)) * HS_ +
                ks + half_id * 8;
            union { h16 v; h8 p[2]; } bu;
            bu.p[0] = *(const h8*)krow;
            bu.p[1] = *(const h8*)(krow + 16);
            h16 bf = bu.v;
            acc[t] = __builtin_amdgcn_wmma_f32_16x16x32_f16(
                false, a.v, false, bf, (short)0, acc[t], false, false);
        }
    }

#pragma unroll
    for (int t = 0; t < 4; ++t) {
        int  col = n0 + t * 16 + ln;
        bool mc  = mask[(size_t)b * S_ + col] != 0;
#pragma unroll
        for (int r = 0; r < 8; ++r) {
            int  row  = r + half_id * 8;
            bool keep = (col >= m0 + row) && mc &&
                        (mask[(size_t)b * S_ + m0 + row] != 0);
            obase[(size_t)row * S_ + col] =
                keep ? acc[t][r] * 0.125f : NEG_MASKED;
        }
    }
}

// ---------------------------------------------------------------------------
extern "C" void kernel_launch(void* const* d_in, const int* in_sizes, int n_in,
                              void* d_out, int out_size, void* d_ws,
                              size_t ws_size, hipStream_t stream) {
    const float*         inputs = (const float*)d_in[0];
    const unsigned char* mask   = (const unsigned char*)d_in[1];  // bool
    const float*         W      = (const float*)d_in[2];
    const float*         bias   = (const float*)d_in[3];
    float*               out    = (float*)d_out;

    // workspace layout (all f16), ~65.3 MB total, fits in L2 (192 MB)
    char* ws = (char*)d_ws;
    _Float16* Ah = (_Float16*)ws; ws += (size_t)M_ * HID_ * 2;           // 12.6 MB
    _Float16* Wt = (_Float16*)ws; ws += (size_t)NQK_ * HID_ * 2;         //  2.4 MB
    _Float16* Hh = (_Float16*)ws; ws += (size_t)M_ * NQK_ * 2;           // 25.2 MB
    _Float16* Qh = (_Float16*)ws; ws += (size_t)M_ * HEADS_ * HS_ * 2;   // 12.6 MB
    _Float16* Kh = (_Float16*)ws;                                        // 12.6 MB

    cvt_inputs<<<(M_ * HID_ + 255) / 256, 256, 0, stream>>>(inputs, Ah,
                                                            M_ * HID_);
    cvt_w_transpose<<<(HID_ * NQK_ + 255) / 256, 256, 0, stream>>>(W, Wt);

    proj_gemm<<<dim3(M_ / 128, NQK_ / 64), 256, 0, stream>>>(Ah, Wt, bias, Hh);

    rope_split<<<(M_ * HEADS_ * (HS_ / 2) + 255) / 256, 256, 0, stream>>>(
        Hh, Qh, Kh);

    attn_logits<<<dim3(B_ * HEADS_, S_ / 16, S_ / 512), 256, 0, stream>>>(
        Qh, Kh, mask, out);
}
